// AutoEncoder_39565238731120
// MI455X (gfx1250) — compile-verified, matrix operand order checked by reading, not run on hardware
//
#include <hip/hip_runtime.h>

typedef __attribute__((ext_vector_type(16))) _Float16 v16h;
typedef __attribute__((ext_vector_type(8)))  _Float16 v8h;
typedef __attribute__((ext_vector_type(8)))  float    v8f;

#define SELU_SCALE 1.0507009873554805f
#define SELU_ALPHA 1.6732632423543772f

__device__ __forceinline__ float selu_f(float x) {
    return SELU_SCALE * (x > 0.f ? x : SELU_ALPHA * (__expf(x) - 1.f));
}
__device__ __forceinline__ float sigmoid_f(float x) {
    return 1.f / (1.f + __expf(-x));
}

// ---------------------------------------------------------------------------
// Weight conversion: f32 (N x K) row-major -> f16 (Np x Kp) row-major, zero pad
// ---------------------------------------------------------------------------
__global__ void convert_pad_f16(const float* __restrict__ src, int N, int K,
                                _Float16* __restrict__ dst, int Np, int Kp) {
    int i = blockIdx.x * blockDim.x + threadIdx.x;
    if (i >= Np * Kp) return;
    int n = i / Kp, k = i % Kp;
    dst[i] = (n < N && k < K) ? (_Float16)src[n * K + k] : (_Float16)0.f;
}

// ---------------------------------------------------------------------------
// Fused conv1(5x5)+bias+maxpool2+selu -> conv2(5x5)+bias+maxpool2+selu
// One image per workgroup, all intermediates in LDS. Output: (B x 320) f16.
// ---------------------------------------------------------------------------
__global__ __launch_bounds__(256) void conv_stack_kernel(
    const float* __restrict__ images,   // B x 1 x 28 x 28
    const float* __restrict__ w1,       // 10 x 1 x 5 x 5
    const float* __restrict__ b1,       // 10
    const float* __restrict__ w2,       // 20 x 10 x 5 x 5
    const float* __restrict__ b2,       // 20
    _Float16* __restrict__ act2)        // B x 320 (c*16 + y*4 + x)
{
    __shared__ float img[784];      // 28*28
    __shared__ float a1[1440];      // 10 x 12 x 12
    __shared__ float w1s[250];
    __shared__ float w2s[5000];
    __shared__ float b1s[10];
    __shared__ float b2s[20];

    const int b   = blockIdx.x;
    const int tid = threadIdx.x;

    for (int i = tid; i < 784; i += 256)  img[i] = images[(size_t)b * 784 + i];
    for (int i = tid; i < 250; i += 256)  w1s[i] = w1[i];
    for (int i = tid; i < 5000; i += 256) w2s[i] = w2[i];
    if (tid < 10) b1s[tid] = b1[tid];
    if (tid >= 32 && tid < 52) b2s[tid - 32] = b2[tid - 32];
    __syncthreads();

    // conv1 + pool + selu : 10 x 12 x 12 outputs
    for (int idx = tid; idx < 1440; idx += 256) {
        int oc = idx / 144;
        int rem = idx % 144;
        int py = rem / 12, px = rem % 12;
        float mx = -3.4e38f;
#pragma unroll
        for (int dy = 0; dy < 2; ++dy)
#pragma unroll
            for (int dx = 0; dx < 2; ++dx) {
                int y = 2 * py + dy, x = 2 * px + dx;  // conv pos in 24x24
                float s = b1s[oc];
#pragma unroll
                for (int ky = 0; ky < 5; ++ky)
#pragma unroll
                    for (int kx = 0; kx < 5; ++kx)
                        s += img[(y + ky) * 28 + (x + kx)] * w1s[oc * 25 + ky * 5 + kx];
                mx = fmaxf(mx, s);
            }
        a1[idx] = selu_f(mx);
    }
    __syncthreads();

    // conv2 + pool + selu : 20 x 4 x 4 outputs, flattened c*16+y*4+x
    for (int idx = tid; idx < 320; idx += 256) {
        int oc = idx / 16;
        int rem = idx % 16;
        int py = rem / 4, px = rem % 4;
        float mx = -3.4e38f;
#pragma unroll
        for (int dy = 0; dy < 2; ++dy)
#pragma unroll
            for (int dx = 0; dx < 2; ++dx) {
                int y = 2 * py + dy, x = 2 * px + dx;  // conv pos in 8x8
                float s = b2s[oc];
                for (int ic = 0; ic < 10; ++ic)
#pragma unroll
                    for (int ky = 0; ky < 5; ++ky)
#pragma unroll
                        for (int kx = 0; kx < 5; ++kx)
                            s += a1[ic * 144 + (y + ky) * 12 + (x + kx)] *
                                 w2s[oc * 250 + ic * 25 + ky * 5 + kx];
                mx = fmaxf(mx, s);
            }
        act2[(size_t)b * 320 + idx] = (_Float16)selu_f(mx);
    }
}

// ---------------------------------------------------------------------------
// WMMA fragment load (ISA 7.12.2, 16-bit A 16x32):
// lane L: row = L&15, g = L>>4, K-base g*8; halves at [g*8 .. g*8+7] and
// [g*8+16 .. g*8+23] -> two 16B loads.
// ---------------------------------------------------------------------------
__device__ __forceinline__ v16h load_frag16x32(const _Float16* __restrict__ p) {
    v8h lo = *(const v8h*)(p);
    v8h hi = *(const v8h*)(p + 16);
    v16h f;
#pragma unroll
    for (int i = 0; i < 8; ++i) { f[i] = lo[i]; f[8 + i] = hi[i]; }
    return f;
}

// ---------------------------------------------------------------------------
// GEMM: C(M x Nvalid) = A(M x K, f16) * W^T, W given as (N x K) f16 row-major.
// One wave (32 threads) per 16x16 output tile. K is compile-time, unrolled.
// ACT: 0 = none, 1 = selu, 2 = sigmoid. OUT_F16 stores f16 with zero padding.
// ---------------------------------------------------------------------------
template <int K, int ACT, bool OUT_F16>
__global__ __launch_bounds__(32) void wmma_gemm_kernel(
    const _Float16* __restrict__ A, int lda,
    const _Float16* __restrict__ W, int ldw,
    const float* __restrict__ bias, int nvalid,
    void* __restrict__ outp, int ldo)
{
    const int lane = threadIdx.x;
    const int row  = lane & 15;
    const int g    = lane >> 4;
    const int mt   = blockIdx.x;
    const int nt   = blockIdx.y;

    const _Float16* ap = A + (size_t)(mt * 16 + row) * lda + g * 8;
    const _Float16* wp = W + (size_t)(nt * 16 + row) * ldw + g * 8;

    v8f acc = {};
#pragma unroll
    for (int k = 0; k < K; k += 32) {
        v16h af = load_frag16x32(ap + k);
        v16h bf = load_frag16x32(wp + k);
        acc = __builtin_amdgcn_wmma_f32_16x16x32_f16(
            false, af, false, bf, (short)0, acc, false, false);
    }

    const int n  = nt * 16 + row;
    const int m0 = mt * 16 + g * 8;
    const float bv = (bias != nullptr && n < nvalid) ? bias[n] : 0.f;

#pragma unroll
    for (int r = 0; r < 8; ++r) {
        float v = acc[r] + bv;
        if (ACT == 1) v = selu_f(v);
        else if (ACT == 2) v = sigmoid_f(v);
        if (OUT_F16) {
            _Float16* o = (_Float16*)outp;
            o[(size_t)(m0 + r) * ldo + n] = (n < nvalid) ? (_Float16)v : (_Float16)0.f;
        } else if (n < nvalid) {
            float* o = (float*)outp;
            o[(size_t)(m0 + r) * ldo + n] = v;
        }
    }
}

// ---------------------------------------------------------------------------
// Per-sample: A = M*M^T (5x5 sym), Jacobi eigendecomposition, top-2 eigvecs,
// P = Uk*Uk^T -> proj (B x 32 f16, 25 valid). Sign of eigvecs cancels in P.
// ---------------------------------------------------------------------------
__global__ __launch_bounds__(256) void svd_proj_kernel(
    const float* __restrict__ code, _Float16* __restrict__ proj, int Bn)
{
    int b = blockIdx.x * blockDim.x + threadIdx.x;
    if (b >= Bn) return;

    float Mm[5][5];
#pragma unroll
    for (int i = 0; i < 5; ++i)
#pragma unroll
        for (int j = 0; j < 5; ++j)
            Mm[i][j] = code[(size_t)b * 25 + i * 5 + j];

    float A[5][5];
#pragma unroll
    for (int i = 0; i < 5; ++i)
#pragma unroll
        for (int j = 0; j < 5; ++j) {
            float s = 0.f;
#pragma unroll
            for (int k = 0; k < 5; ++k) s += Mm[i][k] * Mm[j][k];
            A[i][j] = s;
        }

    float V[5][5];
#pragma unroll
    for (int i = 0; i < 5; ++i)
#pragma unroll
        for (int j = 0; j < 5; ++j) V[i][j] = (i == j) ? 1.f : 0.f;

    for (int sweep = 0; sweep < 10; ++sweep) {
        for (int p = 0; p < 4; ++p) {
            for (int q = p + 1; q < 5; ++q) {
                float apq = A[p][q];
                if (fabsf(apq) < 1e-30f) continue;
                float theta = 0.5f * (A[q][q] - A[p][p]) / apq;
                float t = ((theta >= 0.f) ? 1.f : -1.f) /
                          (fabsf(theta) + sqrtf(theta * theta + 1.f));
                float c = 1.f / sqrtf(t * t + 1.f);
                float s = t * c;
#pragma unroll
                for (int k = 0; k < 5; ++k) {
                    float t1 = A[p][k], t2 = A[q][k];
                    A[p][k] = c * t1 - s * t2;
                    A[q][k] = s * t1 + c * t2;
                }
#pragma unroll
                for (int k = 0; k < 5; ++k) {
                    float t1 = A[k][p], t2 = A[k][q];
                    A[k][p] = c * t1 - s * t2;
                    A[k][q] = s * t1 + c * t2;
                }
#pragma unroll
                for (int k = 0; k < 5; ++k) {
                    float t1 = V[k][p], t2 = V[k][q];
                    V[k][p] = c * t1 - s * t2;
                    V[k][q] = s * t1 + c * t2;
                }
            }
        }
    }

    float ev[5];
#pragma unroll
    for (int k = 0; k < 5; ++k) ev[k] = A[k][k];
    int i1 = 0;
    for (int k = 1; k < 5; ++k) if (ev[k] > ev[i1]) i1 = k;
    int i2 = (i1 == 0) ? 1 : 0;
    for (int k = 0; k < 5; ++k) if (k != i1 && ev[k] > ev[i2]) i2 = k;

    _Float16* pr = proj + (size_t)b * 32;
#pragma unroll
    for (int i = 0; i < 5; ++i)
#pragma unroll
        for (int j = 0; j < 5; ++j) {
            float pv = V[i][i1] * V[j][i1] + V[i][i2] * V[j][i2];
            pr[i * 5 + j] = (_Float16)pv;
        }
#pragma unroll
    for (int k = 25; k < 32; ++k) pr[k] = (_Float16)0.f;
}

// ---------------------------------------------------------------------------
extern "C" void kernel_launch(void* const* d_in, const int* in_sizes, int n_in,
                              void* d_out, int out_size, void* d_ws, size_t ws_size,
                              hipStream_t stream)
{
    (void)in_sizes; (void)n_in; (void)out_size; (void)ws_size;
    const int Bn = 16384;

    const float* images  = (const float*)d_in[0];
    const float* conv1_w = (const float*)d_in[1];
    const float* conv1_b = (const float*)d_in[2];
    const float* conv2_w = (const float*)d_in[3];
    const float* conv2_b = (const float*)d_in[4];
    const float* lin1_w  = (const float*)d_in[5];
    const float* lin1_b  = (const float*)d_in[6];
    const float* lin2_w  = (const float*)d_in[7];
    const float* lin2_b  = (const float*)d_in[8];
    const float* dec1_w  = (const float*)d_in[9];
    const float* dec1_b  = (const float*)d_in[10];
    const float* dec2_w  = (const float*)d_in[11];
    const float* dec2_b  = (const float*)d_in[12];

    float* out  = (float*)d_out;                    // B x 784
    float* code = out + (size_t)Bn * 784;           // B x 25

    char* w = (char*)d_ws;
    _Float16* act2   = (_Float16*)w; w += (size_t)Bn * 320 * 2;  // B x 320
    _Float16* h1     = (_Float16*)w; w += (size_t)Bn * 64  * 2;  // B x 64 (50 valid)
    _Float16* proj   = (_Float16*)w; w += (size_t)Bn * 32  * 2;  // B x 32 (25 valid)
    _Float16* h2     = (_Float16*)w; w += (size_t)Bn * 160 * 2;  // B x 160
    _Float16* w_lin1 = (_Float16*)w; w += (size_t)64  * 320 * 2; // 64 x 320
    _Float16* w_lin2 = (_Float16*)w; w += (size_t)32  * 64  * 2; // 32 x 64
    _Float16* w_dec1 = (_Float16*)w; w += (size_t)160 * 32  * 2; // 160 x 32
    _Float16* w_dec2 = (_Float16*)w; w += (size_t)784 * 160 * 2; // 784 x 160

    // Weight f32 -> f16 with zero padding (N -> mult16, K -> mult32).
    convert_pad_f16<<<(64 * 320 + 255) / 256, 256, 0, stream>>>(lin1_w, 50, 320, w_lin1, 64, 320);
    convert_pad_f16<<<(32 * 64 + 255) / 256, 256, 0, stream>>>(lin2_w, 25, 50, w_lin2, 32, 64);
    convert_pad_f16<<<(160 * 32 + 255) / 256, 256, 0, stream>>>(dec1_w, 160, 25, w_dec1, 160, 32);
    convert_pad_f16<<<(784 * 160 + 255) / 256, 256, 0, stream>>>(dec2_w, 784, 160, w_dec2, 784, 160);

    // Fused conv stack -> act2 (B x 320 f16).
    conv_stack_kernel<<<Bn, 256, 0, stream>>>(images, conv1_w, conv1_b,
                                              conv2_w, conv2_b, act2);

    // lin1: (B x 320) x (320 x 50) + bias, selu -> h1 f16 (B x 64)
    {
        dim3 grid(Bn / 16, 64 / 16);
        wmma_gemm_kernel<320, 1, true><<<grid, 32, 0, stream>>>(
            act2, 320, w_lin1, 320, lin1_b, 50, (void*)h1, 64);
    }
    // lin2: (B x 64) x (64 x 25) + bias -> code f32 (B x 25, into d_out)
    {
        dim3 grid(Bn / 16, 32 / 16);
        wmma_gemm_kernel<64, 0, false><<<grid, 32, 0, stream>>>(
            h1, 64, w_lin2, 64, lin2_b, 25, (void*)code, 25);
    }
    // batched 5x5 SVD projection -> proj f16 (B x 32)
    svd_proj_kernel<<<Bn / 256, 256, 0, stream>>>(code, proj, Bn);

    // dec1: (B x 32) x (32 x 160) + bias, selu -> h2 f16 (B x 160)
    {
        dim3 grid(Bn / 16, 160 / 16);
        wmma_gemm_kernel<32, 1, true><<<grid, 32, 0, stream>>>(
            proj, 32, w_dec1, 32, dec1_b, 160, (void*)h2, 160);
    }
    // dec2: (B x 160) x (160 x 784) + bias, sigmoid -> out f32 (B x 784)
    {
        dim3 grid(Bn / 16, 784 / 16);
        wmma_gemm_kernel<160, 2, false><<<grid, 32, 0, stream>>>(
            h2, 160, w_dec2, 160, dec2_b, 784, (void*)out, 784);
    }
}
